// L_grammar_48601849921777
// MI455X (gfx1250) — matrix operand order repaired and checked by reading, not compiled, not run on hardware
//
#include <hip/hip_runtime.h>
#include <hip/hip_bf16.h>

#define N 256

typedef __bf16 v16bf __attribute__((ext_vector_type(16)));
typedef float  v8f   __attribute__((ext_vector_type(8)));
typedef float  v2f   __attribute__((ext_vector_type(2)));

// ---------------------------------------------------------------------------
// Kernel 0: colsum[j] = sum_i ( W[i,j] + cl_bias[i*N+j] )
// ---------------------------------------------------------------------------
__global__ __launch_bounds__(256) void k_colsum(const float* __restrict__ W,
                                                const float* __restrict__ clb,
                                                float* __restrict__ colsum) {
    int j = threadIdx.x;
    float s = 0.f;
    for (int i = 0; i < N; ++i) s += W[i * N + j] + clb[i * N + j];
    colsum[j] = s;
}

// ---------------------------------------------------------------------------
// Kernel 1: the big GEMM  adjust[b,(m,n)] = context[b,:] . cl_weight[(m,n),:]
// fused with both reductions (never materialized).
// grid = (16 b-tiles, 16 m-tiles), block = 256 (8 waves).
// Wave w handles m = mtile*16 + 2w + {0,1}, all 256 n.
// Outputs: u_adj[b,m] = sum_n adjust[b,m,n]*X[b,n]          (exact per wave)
//          s_part[mtile][b,n] = sum_{m in mtile} adjust[b,m,n] (per-WG partial)
// ---------------------------------------------------------------------------
__global__ __launch_bounds__(256) void k_big(const float* __restrict__ ctx,
                                             const float* __restrict__ clw,
                                             const float* __restrict__ X,
                                             float* __restrict__ s_part,
                                             float* __restrict__ u_adj) {
    const int btile = blockIdx.x;      // 0..15
    const int mtile = blockIdx.y;      // 0..15
    const int b0    = btile * 16;
    const int tid   = threadIdx.x;
    const int wave  = tid >> 5;
    const int lane  = tid & 31;
    const int lo    = lane & 15;
    const int hi    = lane >> 4;

    __shared__ float s_tile[16 * N];   // 16 KB: s partial for this WG's 16 m
    for (int i = tid; i < 16 * N; i += 256) s_tile[i] = 0.f;
    __syncthreads();

    // --- A operand: context tile [16 b x 256 K] as bf16, 8 chunks of K=32.
    // bf16 A 16x32 layout: lane<16 -> elems0..7 = K 0..7, elems8..15 = K16..23;
    // lane>=16 -> K 8..15 / K 24..31.  (kbase = c*32 + hi*8)
    v16bf a[8];
    {
        const float* crow = ctx + (size_t)(b0 + lo) * N;
        #pragma unroll
        for (int c = 0; c < 8; ++c) {
            const int kbase = c * 32 + hi * 8;
            const float4 f0 = *(const float4*)(crow + kbase);
            const float4 f1 = *(const float4*)(crow + kbase + 4);
            const float4 f2 = *(const float4*)(crow + kbase + 16);
            const float4 f3 = *(const float4*)(crow + kbase + 20);
            a[c][0]=(__bf16)f0.x; a[c][1]=(__bf16)f0.y; a[c][2]=(__bf16)f0.z; a[c][3]=(__bf16)f0.w;
            a[c][4]=(__bf16)f1.x; a[c][5]=(__bf16)f1.y; a[c][6]=(__bf16)f1.z; a[c][7]=(__bf16)f1.w;
            a[c][8]=(__bf16)f2.x; a[c][9]=(__bf16)f2.y; a[c][10]=(__bf16)f2.z; a[c][11]=(__bf16)f2.w;
            a[c][12]=(__bf16)f3.x; a[c][13]=(__bf16)f3.y; a[c][14]=(__bf16)f3.z; a[c][15]=(__bf16)f3.w;
        }
    }

    float uprod0[8], uprod1[8];
    #pragma unroll
    for (int v = 0; v < 8; ++v) { uprod0[v] = 0.f; uprod1[v] = 0.f; }

    const int m_0 = mtile * 16 + wave * 2;

    for (int nt = 0; nt < 16; ++nt) {
        // X in D layout: Xt[v] = X[b0+v+8*hi, nt*16+lo]
        float Xt[8];
        #pragma unroll
        for (int v = 0; v < 8; ++v)
            Xt[v] = X[(size_t)(b0 + v + 8 * hi) * N + nt * 16 + lo];

        #pragma unroll
        for (int mm = 0; mm < 2; ++mm) {
            const int m = m_0 + mm;
            v8f acc = {0.f, 0.f, 0.f, 0.f, 0.f, 0.f, 0.f, 0.f};
            // B operand column (mn) = cl_weight row; lane<16: K 0..7/16..23,
            // lane>=16: K 8..15/24..31  => base offset hi*8 within the row.
            const float* brow =
                clw + (size_t)(m * N + nt * 16 + lo) * N + hi * 8;
            #pragma unroll
            for (int c = 0; c < 8; ++c) {
                v16bf bmat;
                #pragma unroll
                for (int q = 0; q < 4; ++q) {
                    const float4 f = *(const float4*)(brow + c * 32 + q * 4 + (q >> 1) * 8);
                    bmat[4*q+0]=(__bf16)f.x; bmat[4*q+1]=(__bf16)f.y;
                    bmat[4*q+2]=(__bf16)f.z; bmat[4*q+3]=(__bf16)f.w;
                }
                acc = __builtin_amdgcn_wmma_f32_16x16x32_bf16(
                    false, a[c], false, bmat, (short)0, acc, false, false);
            }
            // D layout: VGPR v, lane: (b0+v+8*hi, nt*16+lo)
            #pragma unroll
            for (int v = 0; v < 8; ++v) {
                const float dv = acc[v];
                if (mm == 0) uprod0[v] += dv * Xt[v];
                else         uprod1[v] += dv * Xt[v];
                atomicAdd(&s_tile[(v + 8 * hi) * N + nt * 16 + lo], dv);
            }
        }
    }

    // Reduce u products across the 16 lanes of each half-wave.
    #pragma unroll
    for (int v = 0; v < 8; ++v) {
        float r0 = uprod0[v], r1 = uprod1[v];
        #pragma unroll
        for (int off = 1; off < 16; off <<= 1) {
            r0 += __shfl_xor(r0, off, 32);
            r1 += __shfl_xor(r1, off, 32);
        }
        if (lo == 0) {  // lane 0 (hi=0 -> b0+v) and lane 16 (hi=1 -> b0+v+8)
            const int b = b0 + v + 8 * hi;
            u_adj[(size_t)b * N + m_0]     = r0;
            u_adj[(size_t)b * N + m_0 + 1] = r1;
        }
    }

    __syncthreads();
    for (int i = tid; i < 16 * N; i += 256) {
        const int row = i >> 8, col = i & 255;
        s_part[(size_t)mtile * N * N + (size_t)(b0 + row) * N + col] = s_tile[i];
    }
}

// ---------------------------------------------------------------------------
// Kernel 2: s[b,j] = colsum[j] + sum_{mtile} s_part[mtile][b,j]
// ---------------------------------------------------------------------------
__global__ __launch_bounds__(256) void k_sreduce(const float* __restrict__ s_part,
                                                 const float* __restrict__ colsum,
                                                 float* __restrict__ s) {
    const int idx = blockIdx.x * 256 + threadIdx.x;  // b*N + j
    const int j = idx & 255;
    float acc = colsum[j];
    for (int mt = 0; mt < 16; ++mt) acc += s_part[(size_t)mt * N * N + idx];
    s[idx] = acc;
}

// ---------------------------------------------------------------------------
// Kernel 3: per 16x16 (b,m) tile, one wave:
//   d    = s @ A                  (f32 WMMA 16x16x4, K=256)
//   uwb  = sum_n X[b,n]*(W[m,n]+cl_bias[m*N+n])   (f32 WMMA 16x16x4)
//   update[b,m]      = scale*( X[b,m]*d - (u_adj + uwb) )
//   new_context[b,m] = context*(t/(t+1)) + X/(t+1)
// ---------------------------------------------------------------------------
__global__ __launch_bounds__(32) void k_combine(const float* __restrict__ s,
                                                const float* __restrict__ Amat,
                                                const float* __restrict__ X,
                                                const float* __restrict__ W,
                                                const float* __restrict__ clb,
                                                const float* __restrict__ u_adj,
                                                const float* __restrict__ ctx,
                                                const float* __restrict__ scale,
                                                const int* __restrict__ tptr,
                                                float* __restrict__ out) {
    const int b0 = blockIdx.x * 16, m0 = blockIdx.y * 16;
    const int lane = threadIdx.x & 31, lo = lane & 15, hi = lane >> 4;

    v8f dacc = {0.f,0.f,0.f,0.f,0.f,0.f,0.f,0.f};
    v8f uacc = {0.f,0.f,0.f,0.f,0.f,0.f,0.f,0.f};

    for (int k0 = 0; k0 < N; k0 += 4) {
        const int kk = k0 + 2 * hi;  // f32 A 16x4: elems = K {2hi, 2hi+1}
        // d-GEMM: A-op = s[b, kk..kk+1], B-op = Amat[kk.., m0+lo]
        v2f a1, b1;
        a1[0] = s[(size_t)(b0 + lo) * N + kk];
        a1[1] = s[(size_t)(b0 + lo) * N + kk + 1];
        b1[0] = Amat[(size_t)kk * N + m0 + lo];
        b1[1] = Amat[(size_t)(kk + 1) * N + m0 + lo];
        dacc = __builtin_amdgcn_wmma_f32_16x16x4_f32(
            false, a1, false, b1, (short)0, dacc, false, false);
        // u-GEMM: A-op = X[b, kk..], B-op[n,mcol] = W[mcol,n]+clb[mcol*N+n]
        v2f a2, b2;
        a2[0] = X[(size_t)(b0 + lo) * N + kk];
        a2[1] = X[(size_t)(b0 + lo) * N + kk + 1];
        b2[0] = W[(size_t)(m0 + lo) * N + kk]     + clb[(size_t)(m0 + lo) * N + kk];
        b2[1] = W[(size_t)(m0 + lo) * N + kk + 1] + clb[(size_t)(m0 + lo) * N + kk + 1];
        uacc = __builtin_amdgcn_wmma_f32_16x16x4_f32(
            false, a2, false, b2, (short)0, uacc, false, false);
    }

    const float sc = scale[0];
    const int   t  = tptr[0];
    const float ct = (float)t / (float)(t + 1);
    const float xt = 1.0f / (float)(t + 1);

    #pragma unroll
    for (int v = 0; v < 8; ++v) {
        const int b = b0 + v + 8 * hi;
        const int m = m0 + lo;
        const float xv = X[(size_t)b * N + m];
        const float upd =
            sc * (xv * dacc[v] - (u_adj[(size_t)b * N + m] + uacc[v]));
        out[(size_t)b * N + m] = upd;
        out[(size_t)N * N + (size_t)b * N + m] =
            ctx[(size_t)b * N + m] * ct + xv * xt;
    }
}

// ---------------------------------------------------------------------------
extern "C" void kernel_launch(void* const* d_in, const int* in_sizes, int n_in,
                              void* d_out, int out_size, void* d_ws, size_t ws_size,
                              hipStream_t stream) {
    const float* X     = (const float*)d_in[0];
    const float* ctx   = (const float*)d_in[1];
    const float* W     = (const float*)d_in[2];
    const float* scale = (const float*)d_in[3];
    const float* Amat  = (const float*)d_in[4];
    const float* clw   = (const float*)d_in[5];
    const float* clb   = (const float*)d_in[6];
    const int*   t     = (const int*)d_in[7];
    float* out = (float*)d_out;

    char* ws = (char*)d_ws;
    float* s_part = (float*)ws;                                   // 16*N*N f32 (4 MiB)
    float* u_adj  = (float*)(ws + (size_t)16 * N * N * 4);        // N*N f32
    float* s      = (float*)(ws + (size_t)17 * N * N * 4);        // N*N f32
    float* colsum = (float*)(ws + (size_t)18 * N * N * 4);        // N f32

    k_colsum<<<1, 256, 0, stream>>>(W, clb, colsum);
    k_big<<<dim3(16, 16), 256, 0, stream>>>(ctx, clw, X, s_part, u_adj);
    k_sreduce<<<256, 256, 0, stream>>>(s_part, colsum, s);
    k_combine<<<dim3(16, 16), 32, 0, stream>>>(s, Amat, X, W, clb, u_adj,
                                               ctx, scale, t, out);
}
